// LLaDAEMoESparseEMoEBlock_71751723647281
// MI455X (gfx1250) — compile-verified, matrix operand order checked by reading, not run on hardware
//
#include <hip/hip_runtime.h>
#include <hip/hip_bf16.h>
#include <cstdint>

// Problem sizes (fixed by the reference)
#define BB 2
#define SS 2048
#define HH 2048
#define II 1024
#define EE 8
#define TT (BB * SS)          // 4096 tokens
#define TAU_INV 2.0f          // 1/0.5
#define TILE_M 32             // token tile per block (two WMMA M sub-tiles)

typedef __attribute__((ext_vector_type(16))) __bf16        v16bf;
typedef __attribute__((ext_vector_type(8)))  float         v8f;
typedef __attribute__((ext_vector_type(4)))  unsigned int  v4u;
typedef __attribute__((ext_vector_type(4)))  float         v4f;

union Frag16 { v16bf v; v4u q[2]; };

// pack two f32 into two bf16 (truncation) with one v_perm_b32
__device__ __forceinline__ unsigned int pack_bf16(float lo, float hi) {
    return __builtin_amdgcn_perm(__float_as_uint(hi), __float_as_uint(lo), 0x07060302u);
}

// ---------------------------------------------------------------------------
// Kernel 0: f32 -> bf16 bulk conversion (8 elements / thread / iter).
// ---------------------------------------------------------------------------
__global__ void cvt_bf16(const float* __restrict__ src, unsigned int* __restrict__ dst, int n8) {
    int i      = (int)(blockIdx.x * blockDim.x + threadIdx.x);
    int stride = (int)(gridDim.x * blockDim.x);
    const v4f* s4 = (const v4f*)src;
    v4u*       d4 = (v4u*)dst;
    for (; i < n8; i += stride) {
        v4f a = s4[2 * i], b = s4[2 * i + 1];
        v4u o;
        o.x = pack_bf16(a.x, a.y);
        o.y = pack_bf16(a.z, a.w);
        o.z = pack_bf16(b.x, b.y);
        o.w = pack_bf16(b.z, b.w);
        d4[i] = o;
    }
}

// ---------------------------------------------------------------------------
// Kernel 1: routing. One wave32 per token.
// ---------------------------------------------------------------------------
__global__ void moe_route(const float* __restrict__ x,
                          const float* __restrict__ gn,
                          const float* __restrict__ gw,
                          const float* __restrict__ ngw,
                          int*   __restrict__ cnt,
                          int*   __restrict__ tok,
                          float* __restrict__ wt) {
    int t    = (int)((blockIdx.x * blockDim.x + threadIdx.x) >> 5);
    int lane = (int)(threadIdx.x & 31);
    if (t >= TT) return;

    const float* xr = x + (size_t)t * HH;
    float ao[EE]; float ap[EE];
#pragma unroll
    for (int e = 0; e < EE; ++e) { ao[e] = 0.f; ap[e] = 0.f; }

    for (int h = lane; h < HH; h += 32) {
        float xv = xr[h];
#pragma unroll
        for (int e = 0; e < EE; ++e) {
            ao[e] = fmaf(xv, gw[e * HH + h],  ao[e]);
            ap[e] = fmaf(xv, ngw[e * HH + h], ap[e]);
        }
    }
#pragma unroll
    for (int e = 0; e < EE; ++e) {
        for (int m = 16; m >= 1; m >>= 1) {
            ao[e] += __shfl_xor(ao[e], m, 32);
            ap[e] += __shfl_xor(ap[e], m, 32);
        }
    }

    if (lane == 0) {
        float mx = ao[0];
#pragma unroll
        for (int e = 1; e < EE; ++e) mx = fmaxf(mx, ao[e]);
        float pb[EE]; float sum = 0.f;
#pragma unroll
        for (int e = 0; e < EE; ++e) { pb[e] = __expf(ao[e] - mx); sum += pb[e]; }
        float inv = __builtin_amdgcn_rcpf(sum);
        float gl[EE];
#pragma unroll
        for (int e = 0; e < EE; ++e) {
            pb[e] *= inv;
            gl[e] = (ap[e] + gn[t * EE + e]) * TAU_INV;
        }
        int s0 = 0;
#pragma unroll
        for (int e = 1; e < EE; ++e) if (gl[e] > gl[s0]) s0 = e;
        int s1 = (s0 == 0) ? 1 : 0;
#pragma unroll
        for (int e = 0; e < EE; ++e) if (e != s0 && gl[e] > gl[s1]) s1 = e;

        int p0 = atomicAdd(&cnt[s0], 1);
        tok[s0 * TT + p0] = t; wt[s0 * TT + p0] = pb[s0];
        int p1 = atomicAdd(&cnt[s1], 1);
        tok[s1 * TT + p1] = t; wt[s1 * TT + p1] = pb[s1];
    }
}

// ---------------------------------------------------------------------------
// Kernel 2: grouped expert MLP, bf16 WMMA, async global->LDS staging.
// grid = (TT/TILE_M, EE); block = 256 threads (8 wave32).
// Two 16-row A sub-tiles share every B fragment (halves B traffic).
// LDS: X tile bf16 [32][HH] (128KB) + hact bf16 [32][II] (64KB) + tok/wt.
// ---------------------------------------------------------------------------
__global__ void moe_expert(const unsigned short* __restrict__ xb,   // (T,H) bf16
                           const unsigned short* __restrict__ wgb,  // (E,I,H) bf16
                           const unsigned short* __restrict__ wub,  // (E,I,H) bf16
                           const unsigned short* __restrict__ wdb,  // (E,H,I) bf16
                           const int*   __restrict__ cnt,
                           const int*   __restrict__ tok,
                           const float* __restrict__ wt,
                           float* __restrict__ out) {
    extern __shared__ unsigned short lds[];
    unsigned short* x_lds = lds;                              // [32][HH]
    unsigned short* h_lds = lds + TILE_M * HH;                // [32][II]
    int*   tok_lds = (int*)(lds + TILE_M * HH + TILE_M * II); // [32]
    float* wt_lds  = (float*)(tok_lds + TILE_M);              // [32]

    const int e    = (int)blockIdx.y;
    const int ntok = cnt[e];
    const int base = (int)blockIdx.x * TILE_M;
    if (base >= ntok) return;

    const int tid  = (int)threadIdx.x;
    const int lane = tid & 31;
    const int wid  = tid >> 5;
    const int nvalid = min(TILE_M, ntok - base);

    if (tid < TILE_M) {
        int idx = base + min(tid, nvalid - 1);
        tok_lds[tid] = tok[e * TT + idx];
        wt_lds[tid]  = (tid < nvalid) ? wt[e * TT + idx] : 0.f;
    }
    __syncthreads();

    // Stage X tile via CDNA5 async global->LDS copies (16B per op).
    {
        unsigned lds_base = (unsigned)(uintptr_t)x_lds;       // LDS byte offset
        const int nchunk = TILE_M * HH * 2 / 16;              // 8192 x 16B
        for (int c = tid; c < nchunk; c += (int)blockDim.x) {
            int m   = c >> 8;                                 // 256 chunks / row
            int off = (c & 255) << 4;                         // byte offset in row
            const char* g = (const char*)(xb + (size_t)tok_lds[m] * HH) + off;
            unsigned l = lds_base + ((unsigned)c << 4);
            asm volatile("global_load_async_to_lds_b128 %0, %1, off"
                         :: "v"(l), "v"(g) : "memory");
        }
        asm volatile("s_wait_asynccnt 0" ::: "memory");
    }
    __syncthreads();

    const int m16 = lane & 15;          // A row / B column within 16-tile
    const int kh  = lane >> 4;          // half-wave K phase

    // ---- gate / up GEMMs: [32,II] = X[32,HH] @ W^T ; silu(g)*u -> h_lds
    const size_t eW = (size_t)e * II * HH;
    for (int nt = wid; nt < II / 16; nt += 8) {
        const int n = nt * 16 + m16;
        const unsigned short* rg = wgb + eW + (size_t)n * HH;
        const unsigned short* ru = wub + eW + (size_t)n * HH;
        v8f accg0 = {}; v8f accu0 = {};
        v8f accg1 = {}; v8f accu1 = {};
        for (int kb = 0; kb < HH; kb += 32) {
            const int kbase = kb + kh * 8;
            Frag16 a0, a1, bg, bu;
            a0.q[0] = *(const v4u*)(x_lds + m16 * HH + kbase);
            a0.q[1] = *(const v4u*)(x_lds + m16 * HH + kbase + 16);
            a1.q[0] = *(const v4u*)(x_lds + (m16 + 16) * HH + kbase);
            a1.q[1] = *(const v4u*)(x_lds + (m16 + 16) * HH + kbase + 16);
            bg.q[0] = *(const v4u*)(rg + kbase);
            bg.q[1] = *(const v4u*)(rg + kbase + 16);
            bu.q[0] = *(const v4u*)(ru + kbase);
            bu.q[1] = *(const v4u*)(ru + kbase + 16);
            accg0 = __builtin_amdgcn_wmma_f32_16x16x32_bf16(
                false, a0.v, false, bg.v, (short)0, accg0, false, false);
            accg1 = __builtin_amdgcn_wmma_f32_16x16x32_bf16(
                false, a1.v, false, bg.v, (short)0, accg1, false, false);
            accu0 = __builtin_amdgcn_wmma_f32_16x16x32_bf16(
                false, a0.v, false, bu.v, (short)0, accu0, false, false);
            accu1 = __builtin_amdgcn_wmma_f32_16x16x32_bf16(
                false, a1.v, false, bu.v, (short)0, accu1, false, false);
        }
#pragma unroll
        for (int r = 0; r < 8; ++r) {
            int m = r + (kh ? 8 : 0);
            float g0 = accg0[r], u0 = accu0[r];
            float h0 = g0 * __builtin_amdgcn_rcpf(1.f + __expf(-g0)) * u0;
            h_lds[m * II + n] = (unsigned short)(pack_bf16(h0, h0) & 0xFFFFu);
            float g1 = accg1[r], u1 = accu1[r];
            float h1 = g1 * __builtin_amdgcn_rcpf(1.f + __expf(-g1)) * u1;
            h_lds[(m + 16) * II + n] = (unsigned short)(pack_bf16(h1, h1) & 0xFFFFu);
        }
    }
    __syncthreads();

    // ---- down GEMM: y[32,HH] = hact[32,II] @ Wd^T ; scale; scatter-add.
    const size_t eD = (size_t)e * HH * II;
    for (int nt = wid; nt < HH / 16; nt += 8) {
        const int n = nt * 16 + m16;
        const unsigned short* rd = wdb + eD + (size_t)n * II;
        v8f acc0 = {}; v8f acc1 = {};
        for (int kb = 0; kb < II; kb += 32) {
            const int kbase = kb + kh * 8;
            Frag16 a0, a1, b;
            a0.q[0] = *(const v4u*)(h_lds + m16 * II + kbase);
            a0.q[1] = *(const v4u*)(h_lds + m16 * II + kbase + 16);
            a1.q[0] = *(const v4u*)(h_lds + (m16 + 16) * II + kbase);
            a1.q[1] = *(const v4u*)(h_lds + (m16 + 16) * II + kbase + 16);
            b.q[0]  = *(const v4u*)(rd + kbase);
            b.q[1]  = *(const v4u*)(rd + kbase + 16);
            acc0 = __builtin_amdgcn_wmma_f32_16x16x32_bf16(
                false, a0.v, false, b.v, (short)0, acc0, false, false);
            acc1 = __builtin_amdgcn_wmma_f32_16x16x32_bf16(
                false, a1.v, false, b.v, (short)0, acc1, false, false);
        }
#pragma unroll
        for (int r = 0; r < 8; ++r) {
            int m = r + (kh ? 8 : 0);
            float v0 = acc0[r] * wt_lds[m];
            atomicAdd(&out[(size_t)tok_lds[m] * HH + n], v0);
            float v1 = acc1[r] * wt_lds[m + 16];
            atomicAdd(&out[(size_t)tok_lds[m + 16] * HH + n], v1);
        }
    }
}

// ---------------------------------------------------------------------------
extern "C" void kernel_launch(void* const* d_in, const int* in_sizes, int n_in,
                              void* d_out, int out_size, void* d_ws, size_t ws_size,
                              hipStream_t stream) {
    (void)in_sizes; (void)n_in; (void)ws_size;
    const float* x   = (const float*)d_in[0];   // (B,S,H)
    const float* gn  = (const float*)d_in[1];   // (T,E)
    const float* gw  = (const float*)d_in[2];   // (E,H)
    const float* ngw = (const float*)d_in[3];   // (E,H)
    const float* wg  = (const float*)d_in[4];   // (E,I,H)
    const float* wu  = (const float*)d_in[5];   // (E,I,H)
    const float* wd  = (const float*)d_in[6];   // (E,H,I)
    float* out = (float*)d_out;

    // workspace layout (16B-aligned sections)
    const size_t wBytes = (size_t)EE * II * HH * 2;   // 33.5 MB per weight
    const size_t xBytes = (size_t)TT * HH * 2;        // 16.8 MB
    char* ws = (char*)d_ws;
    unsigned short* wgb = (unsigned short*)(ws);
    unsigned short* wub = (unsigned short*)(ws + wBytes);
    unsigned short* wdb = (unsigned short*)(ws + 2 * wBytes);
    unsigned short* xb  = (unsigned short*)(ws + 3 * wBytes);
    int*   cnt = (int*)(ws + 3 * wBytes + xBytes);
    int*   tok = (int*)((char*)cnt + 256);
    float* wt  = (float*)((char*)tok + (size_t)EE * TT * sizeof(int));

    hipMemsetAsync(d_out, 0, (size_t)out_size * sizeof(float), stream);
    hipMemsetAsync(cnt, 0, 256, stream);

    // f32 -> bf16 pre-conversion (weights + activations)
    const int wN8 = (int)(EE * II * HH / 8);
    const int xN8 = (int)(TT * HH / 8);
    cvt_bf16<<<2048, 256, 0, stream>>>(wg, (unsigned int*)wgb, wN8);
    cvt_bf16<<<2048, 256, 0, stream>>>(wu, (unsigned int*)wub, wN8);
    cvt_bf16<<<2048, 256, 0, stream>>>(wd, (unsigned int*)wdb, wN8);
    cvt_bf16<<<2048, 256, 0, stream>>>(x,  (unsigned int*)xb,  xN8);

    moe_route<<<TT / 8, 256, 0, stream>>>(x, gn, gw, ngw, cnt, tok, wt);

    size_t ldsBytes = (size_t)(TILE_M * HH + TILE_M * II) * sizeof(unsigned short)
                    + TILE_M * (sizeof(int) + sizeof(float));
    dim3 grid(TT / TILE_M, EE);
    moe_expert<<<grid, 256, ldsBytes, stream>>>(xb, wgb, wub, wdb, cnt, tok, wt, out);
}
